// LinearInt4_63574105916113
// MI455X (gfx1250) — 8-bit matrix operand order repaired and checked by reading, compiled, not run on hardware
//
#include <hip/hip_runtime.h>
#include <stdint.h>

typedef int v8i __attribute__((ext_vector_type(8)));

#define M_DIM 8192
#define N_DIM 4096
#define K_DIM 4096

#define BM 128
#define BN 128
#define BK 64
#define LDSS 80

#if defined(__gfx1250__) && __has_builtin(__builtin_amdgcn_global_load_async_to_lds_b128)
#define USE_ASYNC 1
#else
#define USE_ASYNC 0
#endif

__device__ __forceinline__ void wait_async0() {
#if USE_ASYNC
#if __has_builtin(__builtin_amdgcn_s_wait_asynccnt)
    __builtin_amdgcn_s_wait_asynccnt(0);
#else
    asm volatile("s_wait_asynccnt 0x0" ::: "memory");
#endif
#endif
}

#if USE_ASYNC
typedef int v4i_vs __attribute__((vector_size(16)));
typedef __attribute__((address_space(1))) v4i_vs* as1_v4i;
typedef __attribute__((address_space(3))) v4i_vs* as3_v4i;

__device__ __forceinline__ void async_cp16(const int8_t* g, int8_t* l) {
    __builtin_amdgcn_global_load_async_to_lds_b128(
        (as1_v4i)(uintptr_t)g,
        (as3_v4i)(uintptr_t)l,
        0, 0);
}
#endif

__global__ __launch_bounds__(256)
void int8_gemm_wmma_kernel(const int8_t* __restrict__ X,
                           const int8_t* __restrict__ W,
                           const float*  __restrict__ bias,
                           float*        __restrict__ out)
{
    __shared__ __align__(16) int8_t As[2][BM * LDSS];
    __shared__ __align__(16) int8_t Bs[2][BN * LDSS];

    const int tid   = threadIdx.x;
    const int lane  = tid & 31;
    const int wave  = tid >> 5;
    const int waveM = wave & 3;
    const int waveN = wave >> 2;
    const int half  = lane >> 4;
    const int l16   = lane & 15;

    const int blockN = blockIdx.x;
    const int blockM = blockIdx.y;

    const size_t aBase = (size_t)blockM * BM * K_DIM;
    const size_t bBase = (size_t)blockN * BN * K_DIM;

    const int ldRow = tid >> 2;
    const int ldCol = (tid & 3) * 16;

    v8i acc[2][4];
    {
        v8i z = {0, 0, 0, 0, 0, 0, 0, 0};
        #pragma unroll
        for (int i = 0; i < 2; ++i)
            #pragma unroll
            for (int j = 0; j < 4; ++j)
                acc[i][j] = z;
    }

    auto issue_stage = [&](int k, int b) {
        #pragma unroll
        for (int p = 0; p < 2; ++p) {
            const int r = ldRow + p * 64;
#if USE_ASYNC
            async_cp16(X + aBase + (size_t)r * K_DIM + k + ldCol, &As[b][r * LDSS + ldCol]);
            async_cp16(W + bBase + (size_t)r * K_DIM + k + ldCol, &Bs[b][r * LDSS + ldCol]);
#else
            *(int4*)(&As[b][r * LDSS + ldCol]) =
                *(const int4*)(X + aBase + (size_t)r * K_DIM + k + ldCol);
            *(int4*)(&Bs[b][r * LDSS + ldCol]) =
                *(const int4*)(W + bBase + (size_t)r * K_DIM + k + ldCol);
#endif
        }
    };

    auto compute = [&](int b) {
        const int8_t* Ab = As[b];
        const int8_t* Bb = Bs[b];

        v8i aF[2];
        #pragma unroll
        for (int tm = 0; tm < 2; ++tm) {
            const int row = waveM * 32 + tm * 16 + l16;
            const int8_t* base = Ab + row * LDSS + half * 16;
            #pragma unroll
            for (int c = 0; c < 2; ++c) {
                int4 t = *(const int4*)(base + c * 32);
                aF[tm][4 * c]     = t.x;
                aF[tm][4 * c + 1] = t.y;
                aF[tm][4 * c + 2] = t.z;
                aF[tm][4 * c + 3] = t.w;
            }
        }

        v8i bF[4];
        #pragma unroll
        for (int tn = 0; tn < 4; ++tn) {
            const int col = waveN * 64 + tn * 16 + l16;
            const int8_t* base = Bb + col * LDSS + half * 16;
            #pragma unroll
            for (int c = 0; c < 2; ++c) {
                int4 t = *(const int4*)(base + c * 32);
                bF[tn][4 * c]     = t.x;
                bF[tn][4 * c + 1] = t.y;
                bF[tn][4 * c + 2] = t.z;
                bF[tn][4 * c + 3] = t.w;
            }
        }

        #pragma unroll
        for (int tm = 0; tm < 2; ++tm)
            #pragma unroll
            for (int tn = 0; tn < 4; ++tn)
                acc[tm][tn] = __builtin_amdgcn_wmma_i32_16x16x64_iu8(
                     true, aF[tm],
                     true, bF[tn],
                    acc[tm][tn],
                     false,  false);
    };

    issue_stage(0, 0);

    for (int k0 = 0; k0 < K_DIM; k0 += 2 * BK) {
        wait_async0();
        __syncthreads();
        issue_stage(k0 + BK, 1);
        compute(0);

        wait_async0();
        __syncthreads();
        if (k0 + 2 * BK < K_DIM)
            issue_stage(k0 + 2 * BK, 0);
        compute(1);
    }

    #pragma unroll
    for (int tm = 0; tm < 2; ++tm) {
        const int row0 = blockM * BM + waveM * 32 + tm * 16 + half * 8;
        #pragma unroll
        for (int tn = 0; tn < 4; ++tn) {
            const int col = blockN * BN + waveN * 64 + tn * 16 + l16;
            const float bv = bias[col];
            #pragma unroll
            for (int r = 0; r < 8; ++r) {
                out[(size_t)(row0 + r) * N_DIM + col] = (float)acc[tm][tn][r] + bv;
            }
        }
    }
}

extern "C" void kernel_launch(void* const* d_in, const int* in_sizes, int n_in,
                              void* d_out, int out_size, void* d_ws, size_t ws_size,
                              hipStream_t stream) {
    const int8_t* x    = (const int8_t*)d_in[0];
    const int8_t* w    = (const int8_t*)d_in[1];
    const float*  bias = (const float*)d_in[2];
    float*        out  = (float*)d_out;

    dim3 grid(N_DIM / BN, M_DIM / BM);
    dim3 block(256);
    int8_gemm_wmma_kernel<<<grid, block, 0, stream>>>(x, w, bias, out);
}
